// GLGAEGradEOL_88330297409972
// MI455X (gfx1250) — matrix operand order repaired and checked by reading, not compiled, not run on hardware
//
#include <hip/hip_runtime.h>
#include <hip/hip_bf16.h>

// ---------------------------------------------------------------------------
// Problem constants (fixed by the reference)
// ---------------------------------------------------------------------------
#define BB 8192
#define GG 1024
#define EE 256
#define KK 2048
#define P_NULL 0.1f

typedef __bf16 bf16_t;
typedef __attribute__((ext_vector_type(16))) __bf16 v16bf;
typedef __attribute__((ext_vector_type(8)))  __bf16 v8bf;
typedef __attribute__((ext_vector_type(8)))  float  v8f;

// ---------------------------------------------------------------------------
// Helpers
// ---------------------------------------------------------------------------
__device__ __forceinline__ bf16_t f2bf(float f) {
    union { float f; unsigned u; } v; v.f = f;
    unsigned r = v.u + 0x7FFFu + ((v.u >> 16) & 1u);   // round-to-nearest-even
    unsigned short h = (unsigned short)(r >> 16);
    return __builtin_bit_cast(bf16_t, h);
}

__device__ __forceinline__ v16bf cat8(v8bf lo, v8bf hi) {
    return __builtin_shufflevector(lo, hi, 0,1,2,3,4,5,6,7,8,9,10,11,12,13,14,15);
}

__device__ __forceinline__ float waveReduceSum(float v) {
    #pragma unroll
    for (int off = 16; off; off >>= 1) v += __shfl_xor(v, off, 32);
    return v;
}
__device__ __forceinline__ float waveReduceMax(float v) {
    #pragma unroll
    for (int off = 16; off; off >>= 1) v = fmaxf(v, __shfl_xor(v, off, 32));
    return v;
}
// 256-thread block reductions (8 waves)
__device__ __forceinline__ float blockReduceSum(float v, float* sm) {
    const int lane = threadIdx.x & 31, wave = threadIdx.x >> 5;
    v = waveReduceSum(v);
    if (lane == 0) sm[wave] = v;
    __syncthreads();
    float r = 0.f;
    #pragma unroll
    for (int w = 0; w < 8; ++w) r += sm[w];
    __syncthreads();
    return r;
}
__device__ __forceinline__ float blockReduceMax(float v, float* sm) {
    const int lane = threadIdx.x & 31, wave = threadIdx.x >> 5;
    v = waveReduceMax(v);
    if (lane == 0) sm[wave] = v;
    __syncthreads();
    float r = -3.402823466e38f;
    #pragma unroll
    for (int w = 0; w < 8; ++w) r = fmaxf(r, sm[w]);
    __syncthreads();
    return r;
}

// ---------------------------------------------------------------------------
// WMMA bf16 GEMM:
//   C[M][N] (f32 row-major) = epi( A[M][Kd] (bf16 rm) * Bt[N][Kd]^T (bf16) )
//   epilogue (if rowbias != nullptr): out = scale*acc - rowbias[m] - colbias[n]
// Block: 256 threads = 8 waves (4 in M x 2 in N), wave tile 32x64 (2x4 WMMA
// 16x16 tiles), block tile 128x128, K step 32. M,N multiples of 128; Kd of 32.
// ---------------------------------------------------------------------------
__global__ __launch_bounds__(256)
void gemm_bf16_wmma(const bf16_t* __restrict__ A, const bf16_t* __restrict__ Bt,
                    float* __restrict__ C, int M, int N, int Kd,
                    float scale, const float* __restrict__ rowbias,
                    const float* __restrict__ colbias)
{
    const int lane  = threadIdx.x & 31;
    const int wave  = threadIdx.x >> 5;
    const int waveM = wave >> 1;              // 0..3
    const int waveN = wave & 1;               // 0..1
    const int blockRow = blockIdx.y * 128;
    const int blockCol = blockIdx.x * 128;

    const int rhalf = lane >> 4;              // 0/1 (lane half)
    const int l15   = lane & 15;

    v8f acc[2][4];
    const v8f vzero = {0.f,0.f,0.f,0.f,0.f,0.f,0.f,0.f};
    #pragma unroll
    for (int mi = 0; mi < 2; ++mi)
        #pragma unroll
        for (int ni = 0; ni < 4; ++ni) acc[mi][ni] = vzero;

    // A fragment base pointers (ISA 16-bit A layout: lane -> row, K split 8+8)
    const bf16_t* Arow[2];
    #pragma unroll
    for (int mi = 0; mi < 2; ++mi)
        Arow[mi] = A + (size_t)(blockRow + waveM*32 + mi*16 + l15) * Kd + rhalf*8;
    // B fragment base pointers (Bt row-major [N][Kd]: lane -> column, K split
    // 8+8 like A: lane half h holds K 8h..8h+7 and 16+8h..16+8h+7)
    const bf16_t* Brow[4];
    #pragma unroll
    for (int ni = 0; ni < 4; ++ni)
        Brow[ni] = Bt + (size_t)(blockCol + waveN*64 + ni*16 + l15) * Kd + rhalf*8;

    for (int k0 = 0; k0 < Kd; k0 += 32) {
        v16bf afrag[2], bfrag[4];
        #pragma unroll
        for (int mi = 0; mi < 2; ++mi)
            afrag[mi] = cat8(*(const v8bf*)(Arow[mi] + k0),
                             *(const v8bf*)(Arow[mi] + k0 + 16));
        #pragma unroll
        for (int ni = 0; ni < 4; ++ni)
            bfrag[ni] = cat8(*(const v8bf*)(Brow[ni] + k0),
                             *(const v8bf*)(Brow[ni] + k0 + 16));
        // prefetch two K-steps ahead (speculative; harmless OOB)
        if (k0 + 64 < Kd) {
            #pragma unroll
            for (int mi = 0; mi < 2; ++mi) __builtin_prefetch(Arow[mi] + k0 + 64, 0, 1);
            #pragma unroll
            for (int ni = 0; ni < 4; ++ni) __builtin_prefetch(Brow[ni] + k0 + 64, 0, 1);
        }
        #pragma unroll
        for (int mi = 0; mi < 2; ++mi)
            #pragma unroll
            for (int ni = 0; ni < 4; ++ni)
                acc[mi][ni] = __builtin_amdgcn_wmma_f32_16x16x32_bf16(
                    false, afrag[mi], false, bfrag[ni],
                    (short)0, acc[mi][ni], false, false);
    }

    // Store (ISA 32-bit C/D layout: lane -> column, 8 rows per lane half)
    #pragma unroll
    for (int mi = 0; mi < 2; ++mi) {
        const int row0 = blockRow + waveM*32 + mi*16 + rhalf*8;
        #pragma unroll
        for (int ni = 0; ni < 4; ++ni) {
            const int col = blockCol + waveN*64 + ni*16 + l15;
            const float cb = colbias ? colbias[col] : 0.f;
            #pragma unroll
            for (int i = 0; i < 8; ++i) {
                float v = acc[mi][ni][i];
                if (rowbias) v = scale*v - rowbias[row0 + i] - cb;
                C[(size_t)(row0 + i) * N + col] = v;
            }
        }
    }
}

// ---------------------------------------------------------------------------
// Memory-bound helper kernels
// ---------------------------------------------------------------------------
__global__ __launch_bounds__(256)
void f2bf_kernel(const float* __restrict__ in, bf16_t* __restrict__ out, size_t n) {
    size_t i = (size_t)blockIdx.x * 256 + threadIdx.x;
    if (i < n) out[i] = f2bf(in[i]);
}

// out[c][r] = bf16(in[r][c])   in: R x C row-major, out: C x R row-major
__global__ __launch_bounds__(256)
void transpose_bf_kernel(const float* __restrict__ in, bf16_t* __restrict__ out,
                         int R, int C) {
    size_t o = (size_t)blockIdx.x * 256 + threadIdx.x;
    if (o >= (size_t)R * C) return;
    int r = (int)(o % (size_t)R);
    int c = (int)(o / (size_t)R);
    out[o] = f2bf(in[(size_t)r * C + c]);
}

// out[k] = mean_r in[r][k]^2   (in: R x C row-major)
__global__ __launch_bounds__(256)
void colmeansq_kernel(const float* __restrict__ in, float* __restrict__ out,
                      int R, int C) {
    int k = blockIdx.x * 256 + threadIdx.x;
    if (k >= C) return;
    float s = 0.f;
    for (int r = 0; r < R; ++r) { float v = in[(size_t)r * C + k]; s += v * v; }
    out[k] = s / (float)R;
}

// Per-row: mean of squares + bf16 copy.  One 256-thread block per row.
__global__ __launch_bounds__(256)
void rowstats_kernel(const float* __restrict__ in, float* __restrict__ mean2,
                     bf16_t* __restrict__ outbf, int C) {
    __shared__ float sm[8];
    const size_t base = (size_t)blockIdx.x * C;
    float s = 0.f;
    for (int i = threadIdx.x; i < C; i += 256) {
        float v = in[base + i];
        s += v * v;
        outbf[base + i] = f2bf(v);
    }
    s = blockReduceSum(s, sm);
    if (threadIdx.x == 0) mean2[blockIdx.x] = s / (float)C;
}

// Row softmax over K=2048; writes f32 and bf16.  One block per row.
__global__ __launch_bounds__(256)
void softmax_kernel(const float* __restrict__ logits, float* __restrict__ p,
                    bf16_t* __restrict__ pbf, int K) {
    __shared__ float sm[8];
    const size_t base = (size_t)blockIdx.x * K;
    float vals[8];
    float mx = -3.402823466e38f;
    #pragma unroll
    for (int i = 0; i < 8; ++i) {
        vals[i] = logits[base + threadIdx.x + i * 256];
        mx = fmaxf(mx, vals[i]);
    }
    mx = blockReduceMax(mx, sm);
    float s = 0.f;
    #pragma unroll
    for (int i = 0; i < 8; ++i) { vals[i] = __expf(vals[i] - mx); s += vals[i]; }
    s = blockReduceSum(s, sm);
    const float inv = 1.f / s;
    #pragma unroll
    for (int i = 0; i < 8; ++i) {
        size_t idx = base + threadIdx.x + i * 256;
        float w = vals[i] * inv;
        p[idx]   = w;
        pbf[idx] = f2bf(w);
    }
}

// EM row update: xq = exp(logit); xqn = xq/(P_NULL+sum xq);
//   use_pik=1: w = pik*xqn; out = bf16(w / sum w)   (the xp operand)
//   use_pik=0: out = bf16(xqn)                      (the decode operand)
__global__ __launch_bounds__(256)
void update_kernel(const float* __restrict__ logits, const float* __restrict__ pik,
                   bf16_t* __restrict__ outbf, int K, int use_pik) {
    __shared__ float sm[8];
    const size_t base = (size_t)blockIdx.x * K;
    float q[8];
    float s = 0.f;
    #pragma unroll
    for (int i = 0; i < 8; ++i) {
        q[i] = __expf(logits[base + threadIdx.x + i * 256]);  // logits <= 0
        s += q[i];
    }
    s = blockReduceSum(s, sm);
    const float inv = 1.f / (P_NULL + s);
    if (use_pik) {
        float t = 0.f;
        #pragma unroll
        for (int i = 0; i < 8; ++i) {
            q[i] = pik[base + threadIdx.x + i * 256] * (q[i] * inv);
            t += q[i];
        }
        t = blockReduceSum(t, sm);
        const float it = 1.f / t;
        #pragma unroll
        for (int i = 0; i < 8; ++i)
            outbf[base + threadIdx.x + i * 256] = f2bf(q[i] * it);
    } else {
        #pragma unroll
        for (int i = 0; i < 8; ++i)
            outbf[base + threadIdx.x + i * 256] = f2bf(q[i] * inv);
    }
}

// Per-row MSE: out[b] = mean_g (recon - images)^2.  One block per row.
__global__ __launch_bounds__(256)
void loss_kernel(const float* __restrict__ recon, const float* __restrict__ images,
                 float* __restrict__ out, int G) {
    __shared__ float sm[8];
    const size_t base = (size_t)blockIdx.x * G;
    float s = 0.f;
    for (int i = threadIdx.x; i < G; i += 256) {
        float d = recon[base + i] - images[base + i];
        s += d * d;
    }
    s = blockReduceSum(s, sm);
    if (threadIdx.x == 0) out[blockIdx.x] = s / (float)G;
}

// ---------------------------------------------------------------------------
// Host orchestration
// ---------------------------------------------------------------------------
extern "C" void kernel_launch(void* const* d_in, const int* in_sizes, int n_in,
                              void* d_out, int out_size, void* d_ws, size_t ws_size,
                              hipStream_t stream) {
    (void)in_sizes; (void)n_in; (void)out_size; (void)ws_size;
    const float* images = (const float*)d_in[0];   // (B, G)
    const float* xa     = (const float*)d_in[1];   // (G, K)
    const float* xb     = (const float*)d_in[2];   // (E, K)
    float* loss = (float*)d_out;                   // (B,)

    // ---- carve workspace -------------------------------------------------
    char* wp = (char*)d_ws;
    auto carve = [&](size_t bytes) -> void* {
        void* r = (void*)wp;
        wp += (bytes + 255) & ~(size_t)255;
        return r;
    };
    float*  lq    = (float*) carve(sizeof(float)  * (size_t)BB * KK); // logits (B,K)
    float*  pik   = (float*) carve(sizeof(float)  * (size_t)BB * KK); // pik f32 (B,K)
    bf16_t* pxbf  = (bf16_t*)carve(sizeof(bf16_t) * (size_t)BB * KK); // pik/xp/xq bf16
    bf16_t* imbf  = (bf16_t*)carve(sizeof(bf16_t) * (size_t)BB * GG); // images bf16
    float*  z     = (float*) carve(sizeof(float)  * (size_t)BB * EE); // z (B,E)
    bf16_t* zbf   = (bf16_t*)carve(sizeof(bf16_t) * (size_t)BB * EE); // z bf16
    bf16_t* xabf  = (bf16_t*)carve(sizeof(bf16_t) * (size_t)GG * KK); // xa  [G][K]
    bf16_t* xaTbf = (bf16_t*)carve(sizeof(bf16_t) * (size_t)GG * KK); // xa^T [K][G]
    bf16_t* xbbf  = (bf16_t*)carve(sizeof(bf16_t) * (size_t)EE * KK); // xb  [E][K]
    bf16_t* xbTbf = (bf16_t*)carve(sizeof(bf16_t) * (size_t)EE * KK); // xb^T [K][E]
    float*  x2    = (float*) carve(sizeof(float) * BB);               // mean images^2
    float*  z2    = (float*) carve(sizeof(float) * BB);               // mean z^2
    float*  c2a   = (float*) carve(sizeof(float) * KK);               // mean_g xa^2
    float*  c2b   = (float*) carve(sizeof(float) * KK);               // mean_e xb^2
    float*  recon = lq;  // recon (B,G) reuses lq buffer (free by then; 33.5MB<=67MB)

    const dim3 blk(256);
    auto gemmGrid = [](int M, int N) { return dim3((unsigned)(N / 128), (unsigned)(M / 128)); };

    // ---- prep: conversions + norm terms ---------------------------------
    rowstats_kernel<<<BB, blk, 0, stream>>>(images, x2, imbf, GG);
    f2bf_kernel<<<(unsigned)(((size_t)GG * KK + 255) / 256), blk, 0, stream>>>(xa, xabf, (size_t)GG * KK);
    transpose_bf_kernel<<<(unsigned)(((size_t)GG * KK + 255) / 256), blk, 0, stream>>>(xa, xaTbf, GG, KK);
    f2bf_kernel<<<(unsigned)(((size_t)EE * KK + 255) / 256), blk, 0, stream>>>(xb, xbbf, (size_t)EE * KK);
    transpose_bf_kernel<<<(unsigned)(((size_t)EE * KK + 255) / 256), blk, 0, stream>>>(xb, xbTbf, EE, KK);
    colmeansq_kernel<<<KK / 256, blk, 0, stream>>>(xa, c2a, GG, KK);
    colmeansq_kernel<<<KK / 256, blk, 0, stream>>>(xb, c2b, EE, KK);

    // ---- encode ----------------------------------------------------------
    // lpik = (2/G) * images@xa - x2[b] - c2a[k]
    gemm_bf16_wmma<<<gemmGrid(BB, KK), blk, 0, stream>>>(
        imbf, xaTbf, lq, BB, KK, GG, 2.f / GG, x2, c2a);
    softmax_kernel<<<BB, blk, 0, stream>>>(lq, pik, pxbf, KK);
    // z = pik @ xb^T
    gemm_bf16_wmma<<<gemmGrid(BB, EE), blk, 0, stream>>>(
        pxbf, xbbf, z, BB, EE, KK, 1.f, nullptr, nullptr);
    rowstats_kernel<<<BB, blk, 0, stream>>>(z, z2, zbf, EE);

    // ---- EM iterations ---------------------------------------------------
    for (int it = 0; it < 4; ++it) {
        // lq = (2/E) * z@xb - z2[b] - c2b[k]
        gemm_bf16_wmma<<<gemmGrid(BB, KK), blk, 0, stream>>>(
            zbf, xbTbf, lq, BB, KK, EE, 2.f / EE, z2, c2b);
        update_kernel<<<BB, blk, 0, stream>>>(lq, pik, pxbf, KK, 1);
        // z = xp @ xb^T
        gemm_bf16_wmma<<<gemmGrid(BB, EE), blk, 0, stream>>>(
            pxbf, xbbf, z, BB, EE, KK, 1.f, nullptr, nullptr);
        rowstats_kernel<<<BB, blk, 0, stream>>>(z, z2, zbf, EE);
    }

    // ---- decode ----------------------------------------------------------
    gemm_bf16_wmma<<<gemmGrid(BB, KK), blk, 0, stream>>>(
        zbf, xbTbf, lq, BB, KK, EE, 2.f / EE, z2, c2b);
    update_kernel<<<BB, blk, 0, stream>>>(lq, pik, pxbf, KK, 0);  // xq normalized
    // recon = xq @ xa^T  (Bt = xa row-major [G][K])
    gemm_bf16_wmma<<<gemmGrid(BB, GG), blk, 0, stream>>>(
        pxbf, xabf, recon, BB, GG, KK, 1.f, nullptr, nullptr);
    loss_kernel<<<BB, blk, 0, stream>>>(recon, images, loss, GG);
}
